// TensorAttention_88261577932965
// MI455X (gfx1250) — matrix operand order repaired and checked by reading, not compiled, not run on hardware
//
#include <hip/hip_runtime.h>
#include <hip/hip_bf16.h>

// Shapes from the reference
#define B_    16
#define T1_   64
#define T2_   64
#define D_    1024
#define PROJ_ 512
#define XS_STRIDE (D_ + 8)   // bf16 elements; +8 (16B) pad to stagger LDS banks

typedef __attribute__((ext_vector_type(16))) __bf16 v16bf;
typedef __attribute__((ext_vector_type(8)))  __bf16 v8bf;
typedef __attribute__((ext_vector_type(4)))  __bf16 v4bf;
typedef __attribute__((ext_vector_type(8)))  float  v8f;

// Branch-free tanh: prefer the gfx1250 hardware V_TANH_F32 builtin; otherwise
// tanh(x) = 1 - 2/(e^{2x}+1) with hardware exp/rcp (saturates via inf/0, no
// EXEC-mask branches, co-executes with XDL WMMA).
static __device__ __forceinline__ float fast_tanh(float x) {
#if __has_builtin(__builtin_amdgcn_tanhf)
    return __builtin_amdgcn_tanhf(x);
#elif __has_builtin(__builtin_amdgcn_tanh_f32)
    return __builtin_amdgcn_tanh_f32(x);
#else
    float e = __expf(2.0f * x);          // v_exp_f32 (+ mul)
#if __has_builtin(__builtin_amdgcn_rcpf)
    float r = __builtin_amdgcn_rcpf(e + 1.0f);  // v_rcp_f32
#else
    float r = 1.0f / (e + 1.0f);
#endif
    return 1.0f - 2.0f * r;
#endif
}

// ---------------------------------------------------------------------------
// Kernel 0: transpose + downconvert W (D x PROJ, f32) -> Wt (PROJ x D, bf16).
// 2 MB read once; Wt (1 MB) then lives in L2 and is reused by all 1024 WGs.
// ---------------------------------------------------------------------------
__global__ __launch_bounds__(256) void prep_wt(const float* __restrict__ W,
                                               __bf16* __restrict__ Wt) {
    int o = blockIdx.x * 256 + threadIdx.x;  // o = n*1024 + k
    int n = o >> 10;
    int k = o & (D_ - 1);
    Wt[o] = (__bf16)W[k * PROJ_ + n];
}

// ---------------------------------------------------------------------------
// Kernel 1: fused  tanh(x@W)@v -> softmax(T2) -> attn-weighted sum of x.
// One workgroup per (b,t1); 256 threads = 8 waves.
//   wave = mtile(0..3) x nhalf(0..1):
//     mtile -> rows 16*mtile..16*mtile+15 of the T2=64 tile (A / C rows)
//     nhalf -> N-chunks [16*nhalf, 16*nhalf+16) of PROJ/16 = 32 chunks
// ---------------------------------------------------------------------------
__global__ __launch_bounds__(256) void attn_fused(
    const float*  __restrict__ x,     // (B,T1,T2,D) f32
    const __bf16* __restrict__ Wt,    // (PROJ,D) bf16
    const float*  __restrict__ vsc,   // (PROJ,) f32
    float*        __restrict__ out)   // (B,T1,D) f32
{
    __shared__ __bf16 xs[T2_][XS_STRIDE];   // 64 x 1032 bf16 = 132 KB
    __shared__ float  scpart[2][T2_];       // per-N-half score partials
    __shared__ float  attn[T2_];

    const int tid    = threadIdx.x;
    const int wave   = tid >> 5;
    const int lane   = tid & 31;
    const int mtile  = wave & 3;
    const int nhalf  = wave >> 2;
    const int halfid = lane >> 4;   // 0: lanes 0-15, 1: lanes 16-31
    const int lmod   = lane & 15;

    const int bt = blockIdx.x;                       // b*T1 + t1
    const float* xblk = x + (size_t)bt * (T2_ * D_); // 64x1024 f32 tile

    // ---- Stage 1: stream x tile from HBM once, convert f32 -> bf16 in LDS
    for (int it = 0; it < 64; ++it) {
        int i   = tid + 256 * it;        // float4 index, 16384 total
        int row = i >> 8;                // 256 float4 per row
        int c4  = i & 255;
        float4 f = ((const float4*)xblk)[i];
        v4bf p;
        p[0] = (__bf16)f.x; p[1] = (__bf16)f.y;
        p[2] = (__bf16)f.z; p[3] = (__bf16)f.w;
        *(v4bf*)&xs[row][c4 * 4] = p;    // one 8B LDS store
    }
    __syncthreads();

    // ---- Stage 2: WMMA GEMM + tanh + dot(v) fused; proj never materialized
    const int arow = mtile * 16 + lmod;          // A row held by this lane
    const __bf16* xrow = &xs[arow][0];

    float sacc[8];
    #pragma unroll
    for (int r = 0; r < 8; ++r) sacc[r] = 0.0f;

    for (int nc = 0; nc < 16; ++nc) {
        const int ncol0 = (nhalf * 16 + nc) * 16;
        // B fragment base: column (ncol0+lmod), K octets split by lane half
        const __bf16* wbase = Wt + (size_t)(ncol0 + lmod) * D_ + halfid * 8;
        const float vcol = vsc[ncol0 + lmod];

        v8f c = {};
        #pragma unroll 4
        for (int kk = 0; kk < 32; ++kk) {
            const int k0 = kk * 32;

            // A 16x32 bf16 fragment (ISA layout):
            //   lane<16 : K = k0+{0..7}  and k0+{16..23}
            //   lane>=16: K = k0+{8..15} and k0+{24..31}
            const __bf16* ap = xrow + k0 + halfid * 8;
            v8bf alo = *(const v8bf*)(ap);        // 16B aligned LDS load
            v8bf ahi = *(const v8bf*)(ap + 16);
            v16bf a;
            #pragma unroll
            for (int j = 0; j < 8; ++j) { a[j] = alo[j]; a[8 + j] = ahi[j]; }

            // B 32x16 bf16 fragment: col = ncol0+lmod,
            //   lane<16: K=k0+{0..7},{16..23}; lane>=16: K=k0+{8..15},{24..31}
            v8bf blo = *(const v8bf*)(wbase + k0);
            v8bf bhi = *(const v8bf*)(wbase + k0 + 16);
            v16bf b;
            #pragma unroll
            for (int j = 0; j < 8; ++j) { b[j] = blo[j]; b[8 + j] = bhi[j]; }

            c = __builtin_amdgcn_wmma_f32_16x16x32_bf16(
                    false, a, false, b, (short)0, c, false, false);
        }

        // C layout: VGPR r, lane -> M = r + 8*halfid (+16*mtile), N = ncol0+lmod
        #pragma unroll
        for (int r = 0; r < 8; ++r)
            sacc[r] += fast_tanh(c[r]) * vcol;
    }

    // Reduce score partials across the 16 lanes of each half (same M rows)
    #pragma unroll
    for (int r = 0; r < 8; ++r) {
        float s = sacc[r];
        s += __shfl_xor(s, 1);
        s += __shfl_xor(s, 2);
        s += __shfl_xor(s, 4);
        s += __shfl_xor(s, 8);
        sacc[r] = s;
    }
    if (lmod == 0) {
        const int rowbase = mtile * 16 + halfid * 8;
        #pragma unroll
        for (int r = 0; r < 8; ++r)
            scpart[nhalf][rowbase + r] = sacc[r];
    }
    __syncthreads();

    // ---- Stage 3: softmax over T2 (tiny; every thread computes identically)
    float mx = -1e30f;
    for (int i = 0; i < T2_; ++i)
        mx = fmaxf(mx, scpart[0][i] + scpart[1][i]);
    float sum = 0.0f;
    for (int i = 0; i < T2_; ++i)
        sum += __expf(scpart[0][i] + scpart[1][i] - mx);
    if (tid < T2_)
        attn[tid] = __expf(scpart[0][tid] + scpart[1][tid] - mx) / sum;
    __syncthreads();

    // ---- Stage 4: out[d] = sum_t2 attn[t2] * x[t2][d]   (x from LDS)
    float* oblk = out + (size_t)bt * D_;
    for (int d = tid; d < D_; d += 256) {
        float acc = 0.0f;
        #pragma unroll 8
        for (int t2 = 0; t2 < T2_; ++t2)
            acc += attn[t2] * (float)xs[t2][d];
        oblk[d] = acc;
    }
}

// ---------------------------------------------------------------------------
extern "C" void kernel_launch(void* const* d_in, const int* in_sizes, int n_in,
                              void* d_out, int out_size, void* d_ws, size_t ws_size,
                              hipStream_t stream) {
    (void)in_sizes; (void)n_in; (void)out_size; (void)ws_size;
    const float* x = (const float*)d_in[0];   // (16,64,64,1024) f32
    const float* W = (const float*)d_in[1];   // (1024,512) f32
    const float* v = (const float*)d_in[2];   // (512,) f32
    __bf16* Wt = (__bf16*)d_ws;               // 1 MB scratch: (512,1024) bf16

    prep_wt<<<(PROJ_ * D_) / 256, 256, 0, stream>>>(W, Wt);
    attn_fused<<<B_ * T1_, 256, 0, stream>>>(x, Wt, v, (float*)d_out);
}
